// ContextualLoss_complex_76338748719505
// MI455X (gfx1250) — matrix operand order repaired and checked by reading, not compiled, not run on hardware
//
#include <hip/hip_runtime.h>
#include <hip/hip_bf16.h>
#include <math.h>

// ---------------------------------------------------------------------------
// ContextualLoss (CoCosNet), forward-direction, B=4 C=256 H=W=64 (N=4096).
// Pipeline:
//   k1: meanY[b][c]        (spatial mean of Y)
//   k2: Xh/Yh = f16( (X|Y - meanY) / (||.||_2 + eps) ), stored k-major [b][n][C]
//   k3: per 16-row strip: WMMA f32<=f16 GEMM of 16x4096 similarities into a
//       256KB LDS strip (CDNA5 320KB/WGP), then in-LDS row min + exp-sum.
//       128 threads = 4 waves = 1 wave/SIMD: full register-file slice per wave
//       (waves_per_eu(1)) so the fenced 16-fragment dual-tile pipeline fits
//       with no scratch spills; 32 b128 loads in flight under the WMMA block.
//   k4: per-batch deterministic reduction -> loss[b] = -log(mean(maxA))
// ---------------------------------------------------------------------------

typedef __attribute__((ext_vector_type(16))) _Float16 v16h;
typedef __attribute__((ext_vector_type(8)))  _Float16 v8h;
typedef __attribute__((ext_vector_type(8)))  float    v8f;

#define BB 4
#define CC 256
#define NN 4096
#define RS 4100                 // LDS row stride (floats); 8*RS % 64 == 32 -> conflict-free WMMA stores
#define EPSN 2.220446049250313e-16f
#define INV_H 10.0f             // 1 / bandwidth(0.1)

#if defined(__has_builtin)
#if __has_builtin(__builtin_amdgcn_sched_barrier)
#define SCHED_FENCE() __builtin_amdgcn_sched_barrier(0)
#endif
#endif
#ifndef SCHED_FENCE
#define SCHED_FENCE() asm volatile("" ::: "memory")
#endif

// ---------------- k1: per-channel spatial mean of Y ----------------
__global__ void cl_mean_kernel(const float* __restrict__ Y, float* __restrict__ meanY) {
    int bc = blockIdx.x;                       // 0 .. B*C-1
    const float* p = Y + (size_t)bc * NN;
    float s = 0.f;
    for (int i = threadIdx.x; i < NN; i += 256) s += p[i];
    __shared__ float red[256];
    red[threadIdx.x] = s;
    __syncthreads();
    for (int o = 128; o > 0; o >>= 1) {
        if (threadIdx.x < o) red[threadIdx.x] += red[threadIdx.x + o];
        __syncthreads();
    }
    if (threadIdx.x == 0) meanY[bc] = red[0] * (1.0f / NN);
}

// ------- k2: center by meanY, L2-normalize over C, emit f16 k-major -------
__global__ void cl_norm_kernel(const float* __restrict__ X, const float* __restrict__ Y,
                               const float* __restrict__ meanY,
                               _Float16* __restrict__ Xh, _Float16* __restrict__ Yh) {
    int idx = blockIdx.x * 256 + threadIdx.x;  // 0 .. B*N-1
    int b = idx >> 12;
    int n = idx & (NN - 1);
    const float* mb = meanY + b * CC;
    size_t base = (size_t)b * CC * NN + n;
    float ssx = 0.f, ssy = 0.f;
    for (int c = 0; c < CC; c++) {
        float m  = mb[c];
        float xv = X[base + (size_t)c * NN] - m;
        float yv = Y[base + (size_t)c * NN] - m;
        ssx += xv * xv;
        ssy += yv * yv;
    }
    float rx = 1.0f / (sqrtf(ssx) + EPSN);
    float ry = 1.0f / (sqrtf(ssy) + EPSN);
    _Float16* xo = Xh + ((size_t)(b * NN + n)) * CC;
    _Float16* yo = Yh + ((size_t)(b * NN + n)) * CC;
    for (int c = 0; c < CC; c++) {
        float m = mb[c];
        xo[c] = (_Float16)((X[base + (size_t)c * NN] - m) * rx);
        yo[c] = (_Float16)((Y[base + (size_t)c * NN] - m) * ry);
    }
}

// ------- k3: fused WMMA similarity strip + contextual reduction -------
// grid = (N/16, B), block = 128 (4 wave32 -> 1 wave/SIMD), dynamic LDS below.
// 1 workgroup/WGP (LDS-bound) and 1 wave/SIMD, so let each wave own the full
// register-file slice: ~230 live VGPRs (afrag 64 + 16 B-frags 128 + 2 accum
// 16 + addressing) with zero spills and 32 loads in flight.
__global__ void
__attribute__((amdgpu_flat_work_group_size(128, 128), amdgpu_waves_per_eu(1)))
cl_cx_kernel(const _Float16* __restrict__ Xh, const _Float16* __restrict__ Yh,
             float* __restrict__ blockSum) {
    extern __shared__ float smem[];
    float* S       = smem;                 // [16][RS] similarity strip (256 KB)
    float* aux     = S + 16 * RS;          // [16][8] partials
    float* dminArr = aux + 128;            // [16]

    const int nblk = blockIdx.x;           // row strip within batch
    const int b    = blockIdx.y;
    const int n0   = nblk * 16;
    const int t    = threadIdx.x;
    const int lane = t & 31;
    const int wave = t >> 5;               // 0..3
    const int r    = lane & 15;            // row (A) / col (B) within tile
    const int g    = lane >> 4;            // lane group

    // ---- load this wave's A fragments (16 rows x 256 K) into VGPRs ----
    // Layout (matches B), A 16x32, contiguous k order chosen here: lanes 0-15 hold K = 0..15,
    // lanes 16-31 hold K = 16..31, M = lane&15.
    const _Float16* Xrow = Xh + ((size_t)(b * NN + n0 + r)) * CC;
    const int kb0 = g * 16;
    v16h afrag[8];
#pragma unroll
    for (int kk = 0; kk < 8; kk++) {
        const _Float16* p = Xrow + 32 * kk + kb0;
        v8h lo = *(const v8h*)p;
        v8h hi = *(const v8h*)(p + 8);
        v16h a;
#pragma unroll
        for (int i = 0; i < 8; i++) { a[i] = lo[i]; a[8 + i] = hi[i]; }
        afrag[kk] = a;
    }

    // ---- sweep all 256 column tiles; 2 tiles per iteration per wave ----
    // B 32x16, contiguous k order chosen here: lanes 0-15 hold K=0..15, lanes 16-31 hold K=16..31,
    // N = lane&15  -> one contiguous 16-half (32B) chunk per lane per K-step.
    const int kbB = g * 16;
    const _Float16* Ybase = Yh + ((size_t)(b * NN + r)) * CC + kbB;

    for (int tile = wave; tile < NN / 16; tile += 8) {
        const int m0a = tile * 16;                    // tiles: wave + 8*i
        const int m0b = m0a + 64;                     // and    wave + 8*i + 4
        const _Float16* Ya = Ybase + (size_t)m0a * CC;
        const _Float16* Yb = Ybase + (size_t)m0b * CC;

        // prefetch next tile pair's panels toward the WGP (global_prefetch_b8)
        if (tile + 8 < NN / 16) {
            __builtin_prefetch(Ya + (size_t)128 * CC, 0, 3);
            __builtin_prefetch(Yb + (size_t)128 * CC, 0, 3);
        }

        // Phase 1: issue all 32 b128 loads into 16 distinct fragment registers.
        v16h bfA[8], bfB[8];
#pragma unroll
        for (int kk = 0; kk < 8; kk++) {
            bfA[kk] = *(const v16h*)(Ya + 32 * kk);
            bfB[kk] = *(const v16h*)(Yb + 32 * kk);
        }
        // Hard scheduling fence: loads may not sink into the WMMA block, so
        // they keep distinct destinations (no WAR recycling) and the waitcnt
        // pass can use partial s_wait_loadcnt values under the WMMA chain.
        SCHED_FENCE();

        // Phase 2: two independent accumulate chains -> interleaved WMMA issue
        v8f ca = {}, cb = {};
#pragma unroll
        for (int kk = 0; kk < 8; kk++) {
            ca = __builtin_amdgcn_wmma_f32_16x16x32_f16(
                     false, afrag[kk], false, bfA[kk], (short)0, ca, false, false);
            cb = __builtin_amdgcn_wmma_f32_16x16x32_f16(
                     false, afrag[kk], false, bfB[kk], (short)0, cb, false, false);
        }
        SCHED_FENCE();

        // Phase 3: C/D layout: lane group g -> rows j + 8g, cols m0 + (lane&15)
#pragma unroll
        for (int j = 0; j < 8; j++) {
            S[(j + 8 * g) * RS + m0a + r] = ca[j];
            S[(j + 8 * g) * RS + m0b + r] = cb[j];
        }
    }
    __syncthreads();

    // ---- reduction over the 16 x 4096 strip (128 threads: 8 per row) ----
    const int rr = t >> 3;                 // row 0..15
    const int ii = t & 7;                  // 8 threads per row

    // pass 1: row max of similarity  (dmin = 1 - smax)
    float smax = -1e30f;
    for (int col = ii; col < NN; col += 8) smax = fmaxf(smax, S[rr * RS + col]);
    aux[rr * 8 + ii] = smax;
    __syncthreads();
    if (t < 16) {
        float m = aux[t * 8];
        for (int i = 1; i < 8; i++) m = fmaxf(m, aux[t * 8 + i]);
        dminArr[t] = 1.0f - m;
    }
    __syncthreads();

    // pass 2: row sum of exp((1 - d/(dmin+1e-5)) / h)
    float dmin = dminArr[rr];
    float inv  = 1.0f / (dmin + 1e-5f);
    float z = 0.f;
    for (int col = ii; col < NN; col += 8) {
        float d = 1.0f - S[rr * RS + col];
        z += __expf((1.0f - d * inv) * INV_H);
    }
    aux[rr * 8 + ii] = z;
    __syncthreads();
    if (t < 16) {
        float Z = 0.f;
        for (int i = 0; i < 8; i++) Z += aux[t * 8 + i];
        float dm   = dminArr[t];
        float maxA = __expf((1.0f - dm / (dm + 1e-5f)) * INV_H) / Z;  // max_m A[n][m]
        aux[t] = maxA;                                                // reuse aux[0..15]
    }
    __syncthreads();
    if (t == 0) {
        float s = 0.f;
        for (int i = 0; i < 16; i++) s += aux[i];
        blockSum[b * (NN / 16) + nblk] = s;   // deterministic partial (no atomics)
    }
}

// ------- k4: loss[b] = -log(mean_n maxA) -------
__global__ void cl_fin_kernel(const float* __restrict__ blockSum, float* __restrict__ out) {
    int b = blockIdx.x;
    __shared__ float red[256];
    red[threadIdx.x] = blockSum[b * 256 + threadIdx.x];
    __syncthreads();
    for (int o = 128; o > 0; o >>= 1) {
        if (threadIdx.x < o) red[threadIdx.x] += red[threadIdx.x + o];
        __syncthreads();
    }
    if (threadIdx.x == 0) out[b] = -logf(red[0] * (1.0f / NN));
}

extern "C" void kernel_launch(void* const* d_in, const int* in_sizes, int n_in,
                              void* d_out, int out_size, void* d_ws, size_t ws_size,
                              hipStream_t stream) {
    const float* X = (const float*)d_in[0];
    const float* Y = (const float*)d_in[1];
    float* out = (float*)d_out;

    // workspace layout (bytes):
    //   [0,      4096)  meanY    : B*C f32
    //   [4096,   8192)  blockSum : B*256 f32
    //   [8192,  +8MB )  Xh       : B*N*C f16
    //   [ ... , +8MB )  Yh       : B*N*C f16
    char* ws = (char*)d_ws;
    float*    meanY    = (float*)(ws);
    float*    blockSum = (float*)(ws + 4096);
    _Float16* Xh       = (_Float16*)(ws + 8192);
    _Float16* Yh       = (_Float16*)(ws + 8192 + (size_t)BB * NN * CC * sizeof(_Float16));

    const int smem_bytes = (16 * RS + 144) * (int)sizeof(float);   // ~263 KB (<320 KB/WGP)
    hipFuncSetAttribute((const void*)cl_cx_kernel,
                        hipFuncAttributeMaxDynamicSharedMemorySize, smem_bytes);

    cl_mean_kernel<<<BB * CC, 256, 0, stream>>>(Y, meanY);
    cl_norm_kernel<<<(BB * NN) / 256, 256, 0, stream>>>(X, Y, meanY, Xh, Yh);
    cl_cx_kernel<<<dim3(NN / 16, BB), 128, smem_bytes, stream>>>(Xh, Yh, blockSum);
    cl_fin_kernel<<<BB, 256, 0, stream>>>(blockSum, out);
}
